// ModulatedDeformRoIPoolingPack_41463614276038
// MI455X (gfx1250) — matrix operand order repaired and checked by reading, not compiled, not run on hardware
//
#include <hip/hip_runtime.h>
#include <hip/hip_bf16.h>

// ---------------------------------------------------------------------------
// ModulatedDeformRoIPoolingPack for MI455X (gfx1250, wave32, WMMA)
//
// Pipeline:
//   pool1  : deform_roi_pool(data, rois, offset=None)  -> xf   [N, 12544] f16
//   gemm   : h1 = relu(xf @ w1 + b1)                   -> [N,1024]
//   gemm   : h2 = relu(h1 @ w2 + b2)                   -> [N,1024]
//   gemm   : off = h2 @ w3 + b3                        -> [N,98]  (padded 128)
//   gemm   : m1 = relu(xf @ mw1 + mb1)                 -> [N,1024]
//   gemm   : msk = sigmoid(m1 @ mw2 + mb2)             -> [N,49]  (padded 128)
//   pool2  : deform_roi_pool(data, rois, off) * msk    -> out [N,256,7,7] f32
//
// GEMMs: v_wmma_f32_16x16x32_f16, one wave per 16x128 output tile (8 WMMA
// accumulators), all 8 B fragments loaded before the WMMA chain for ILP.
// ---------------------------------------------------------------------------

#define OUTD 7
#define CCH 256
#define SPP 4
#define TRANS_STD 0.1f
#define SCALEF 0.0625f
#define FCD 1024
#define HH 128
#define WW 128
#define NROI 2000
#define FIN (OUTD * OUTD * CCH)          // 12544
#define NSAMP (OUTD * OUTD * SPP * SPP)  // 784
#define NTILE 8                          // 8 x 16 = 128 output cols per wave

typedef _Float16 v16h __attribute__((ext_vector_type(16)));
typedef _Float16 v8h  __attribute__((ext_vector_type(8)));
typedef float    v8f  __attribute__((ext_vector_type(8)));

// ------------------------------ weight prep --------------------------------
// W  : [K, Nreal] fp32 (k-major, as in the reference `xf @ w`)
// WT : [Npad, K] f16   (n-major, zero-padded rows) so each lane's B fragment
//      is one contiguous 32-byte load.
__global__ void cvt_weight(const float* __restrict__ W, _Float16* __restrict__ WT,
                           int K, int Nreal, int Npad) {
  size_t total = (size_t)Npad * (size_t)K;
  for (size_t idx = (size_t)blockIdx.x * blockDim.x + threadIdx.x; idx < total;
       idx += (size_t)gridDim.x * blockDim.x) {
    size_t o = idx / (size_t)K;
    size_t k = idx - o * (size_t)K;
    float v = (o < (size_t)Nreal) ? W[k * (size_t)Nreal + o] : 0.0f;
    WT[idx] = (_Float16)v;
  }
}

__global__ void pad_bias(const float* __restrict__ b, float* __restrict__ bp,
                         int Nreal, int Npad) {
  int i = blockIdx.x * blockDim.x + threadIdx.x;
  if (i < Npad) bp[i] = (i < Nreal) ? b[i] : 0.0f;
}

// ------------------------------ GEMM (WMMA) --------------------------------
// O = act(A @ B^T + bias).  A: [M,K] f16 row-major, BT: [Npad,K] f16.
// One wave per 16x128 output tile; 8 accumulators; K stepped by 32.
// act: 0 = relu, 1 = none, 2 = sigmoid.
__global__ __launch_bounds__(32)
void gemm_wmma(const _Float16* __restrict__ A, const _Float16* __restrict__ BT,
               const float* __restrict__ biasP, float* __restrict__ O32,
               _Float16* __restrict__ O16, int M, int K, int Npad, int act) {
  const int lane  = threadIdx.x;          // 0..31 (wave32)
  const int msel  = lane & 15;
  const int khalf = lane >> 4;            // 0: lanes 0-15, 1: lanes 16-31
  const int m0 = blockIdx.x * 16;
  const int n0 = blockIdx.y * (16 * NTILE);

  const _Float16* arow = A + (size_t)(m0 + msel) * (size_t)K;
  const _Float16* brow[NTILE];
#pragma unroll
  for (int j = 0; j < NTILE; ++j)
    brow[j] = BT + (size_t)(n0 + j * 16 + msel) * (size_t)K;

  v8f acc[NTILE] = {};

  for (int k = 0; k < K; k += 32) {
    // A fragment: 16x32 f16, same k order as B:
    //   lanes 0-15 : K = k..k+15
    //   lanes 16-31: K = k+16..k+31
    v16h afrag = *(const v16h*)(arow + k + khalf * 16);

    // B fragments: 32x16 f16: lanes 0-15 hold K=k..k+15 (N=lane),
    // lanes 16-31 hold K=k+16..k+31 (N=lane-16) -> one 32B load per tile.
    // Load ALL tiles first so the 9 loads issue as one clause ahead of the
    // WMMA chain (single wait, 8 back-to-back WMMAs).
    v16h bfrag[NTILE];
#pragma unroll
    for (int j = 0; j < NTILE; ++j)
      bfrag[j] = *(const v16h*)(brow[j] + k + khalf * 16);

    // Prefetch next K strip of B into cache.
    if (k + 32 < K) {
#pragma unroll
      for (int j = 0; j < NTILE; ++j)
        __builtin_prefetch((const void*)(brow[j] + k + 32), 0, 1);
    }

#pragma unroll
    for (int j = 0; j < NTILE; ++j) {
      acc[j] = __builtin_amdgcn_wmma_f32_16x16x32_f16(
          /*neg_a=*/false, afrag, /*neg_b=*/false, bfrag[j],
          /*c_mod=*/(short)0, acc[j], /*reuse_a=*/false, /*reuse_b=*/false);
    }
  }

  // Epilogue: C/D layout — VGPR r: lanes 0-15 -> M=m0+r, lanes 16-31 -> M=m0+8+r;
  // N = n-tile base + (lane & 15).
#pragma unroll
  for (int j = 0; j < NTILE; ++j) {
    const int col = n0 + j * 16 + msel;
    const float bias = biasP[col];
#pragma unroll
    for (int r = 0; r < 8; ++r) {
      int row = m0 + r + (khalf ? 8 : 0);
      float v = acc[j][r] + bias;
      if (act == 0)      v = fmaxf(v, 0.0f);
      else if (act == 2) v = 1.0f / (1.0f + __expf(-v));
      size_t oi = (size_t)row * (size_t)Npad + (size_t)col;
      O32[oi] = v;
      O16[oi] = (_Float16)v;
    }
  }
}

// ------------------------------ ROI pooling --------------------------------
// stage 0: offset == NULL, writes xf16[n][c*49 + bin]
// stage 1: offsets from off[n][{0,49}+bin], multiply by mask[n][bin],
//          writes out[n][c][bin] (fp32, contiguous [N,256,7,7]).
__global__ __launch_bounds__(CCH)
void pool_kernel(const float* __restrict__ data, const float* __restrict__ rois,
                 const float* __restrict__ off, int off_stride,
                 const float* __restrict__ mask, int mask_stride,
                 _Float16* __restrict__ xf16, float* __restrict__ out, int stage) {
  __shared__ int   sx0[NSAMP], sy0[NSAMP], sx1[NSAMP], sy1[NSAMP];
  __shared__ float sw00[NSAMP], sw01[NSAMP], sw10[NSAMP], sw11[NSAMP];
  __shared__ float svld[NSAMP];
  __shared__ float sinv[OUTD * OUTD];

  const int n = blockIdx.x;
  const int t = threadIdx.x;

  const float r0 = rois[n * 5 + 0];
  const float r1 = rois[n * 5 + 1];
  const float r2 = rois[n * 5 + 2];
  const float r3 = rois[n * 5 + 3];
  const float r4 = rois[n * 5 + 4];
  const int bi = (int)r0;

  const float sw = floorf(r1 + 0.5f) * SCALEF - 0.5f;
  const float sh = floorf(r2 + 0.5f) * SCALEF - 0.5f;
  const float ew = (floorf(r3 + 0.5f) + 1.0f) * SCALEF - 0.5f;
  const float eh = (floorf(r4 + 0.5f) + 1.0f) * SCALEF - 0.5f;
  const float rw = fmaxf(ew - sw, 0.1f);
  const float rh = fmaxf(eh - sh, 0.1f);
  const float bw = rw / (float)OUTD, bh = rh / (float)OUTD;
  const float subw = bw / (float)SPP, subh = bh / (float)SPP;

  // Phase A: sample descriptors (channel independent).
  for (int s = t; s < NSAMP; s += CCH) {
    const int bin = s >> 4;          // /16
    const int sub = s & 15;
    const int ph = bin / OUTD, pw = bin % OUTD;
    const int ih = sub >> 2, iw = sub & 3;

    float wst = (float)pw * bw + sw;
    float hst = (float)ph * bh + sh;
    if (stage) {
      wst += off[(size_t)n * off_stride + bin] * TRANS_STD * rw;
      hst += off[(size_t)n * off_stride + 49 + bin] * TRANS_STD * rh;
    }
    const float w = wst + (float)iw * subw;
    const float h = hst + (float)ih * subh;

    const bool valid = (w >= -0.5f) && (w <= (float)WW - 0.5f) &&
                       (h >= -0.5f) && (h <= (float)HH - 0.5f);
    const float wc = fminf(fmaxf(w, 0.0f), (float)(WW - 1));
    const float hc = fminf(fmaxf(h, 0.0f), (float)(HH - 1));
    const int x0 = (int)floorf(wc);
    const int y0 = (int)floorf(hc);
    const int x1 = min(x0 + 1, WW - 1);
    const int y1 = min(y0 + 1, HH - 1);
    const float dx = wc - (float)x0;
    const float dy = hc - (float)y0;
    const float vf = valid ? 1.0f : 0.0f;

    sx0[s] = x0; sy0[s] = y0; sx1[s] = x1; sy1[s] = y1;
    sw00[s] = (1.0f - dx) * (1.0f - dy) * vf;
    sw01[s] = dx * (1.0f - dy) * vf;
    sw10[s] = (1.0f - dx) * dy * vf;
    sw11[s] = dx * dy * vf;
    svld[s] = vf;
  }
  __syncthreads();

  // Phase B: per-bin sample counts -> reciprocal (0 when cnt==0).
  if (t < OUTD * OUTD) {
    float cnt = 0.0f;
#pragma unroll
    for (int i = 0; i < SPP * SPP; ++i) cnt += svld[t * 16 + i];
    sinv[t] = (cnt > 0.0f) ? (1.0f / fmaxf(cnt, 1.0f)) : 0.0f;
  }
  __syncthreads();

  // Phase C: one thread per channel.
  const int c = t;
  const float* dptr = data + ((size_t)bi * CCH + c) * (size_t)(HH * WW);
  for (int bin = 0; bin < OUTD * OUTD; ++bin) {
    float acc = 0.0f;
#pragma unroll 4
    for (int i = 0; i < SPP * SPP; ++i) {
      const int s = bin * 16 + i;
      const int x0 = sx0[s], y0 = sy0[s], x1 = sx1[s], y1 = sy1[s];
      const float v00 = dptr[y0 * WW + x0];
      const float v01 = dptr[y0 * WW + x1];
      const float v10 = dptr[y1 * WW + x0];
      const float v11 = dptr[y1 * WW + x1];
      acc += sw00[s] * v00 + sw01[s] * v01 + sw10[s] * v10 + sw11[s] * v11;
    }
    const float val = acc * sinv[bin];
    if (stage) {
      const float mk = mask[(size_t)n * mask_stride + bin];
      out[((size_t)n * CCH + c) * (OUTD * OUTD) + bin] = val * mk;
    } else {
      xf16[(size_t)n * FIN + (size_t)c * (OUTD * OUTD) + bin] = (_Float16)val;
    }
  }
}

// ------------------------------ launcher -----------------------------------
extern "C" void kernel_launch(void* const* d_in, const int* in_sizes, int n_in,
                              void* d_out, int out_size, void* d_ws, size_t ws_size,
                              hipStream_t stream) {
  const float* data = (const float*)d_in[0];   // [4,256,128,128]
  const float* rois = (const float*)d_in[1];   // [2000,5]
  const float* w1   = (const float*)d_in[2];   // [12544,1024]
  const float* b1   = (const float*)d_in[3];
  const float* w2   = (const float*)d_in[4];   // [1024,1024]
  const float* b2   = (const float*)d_in[5];
  const float* w3   = (const float*)d_in[6];   // [1024,98]
  const float* b3   = (const float*)d_in[7];
  const float* mw1  = (const float*)d_in[8];   // [12544,1024]
  const float* mb1  = (const float*)d_in[9];
  const float* mw2  = (const float*)d_in[10];  // [1024,49]
  const float* mb2  = (const float*)d_in[11];
  float* out = (float*)d_out;                  // [2000,256,7,7]

  const int NP3 = 128;   // pad 98  -> 128
  const int NPM = 128;   // pad 49  -> 128 (N-block is 128 wide)

  // Workspace carve-out (256B aligned regions).
  size_t off = 0;
  char* base = (char*)d_ws;
  auto alloc = [&](size_t bytes) -> char* {
    off = (off + 255) & ~(size_t)255;
    char* p = base + off;
    off += bytes;
    return p;
  };
  _Float16* xf16  = (_Float16*)alloc((size_t)NROI * FIN * 2);
  _Float16* w1T   = (_Float16*)alloc((size_t)FCD * FIN * 2);
  _Float16* mw1T  = (_Float16*)alloc((size_t)FCD * FIN * 2);
  _Float16* w2T   = (_Float16*)alloc((size_t)FCD * FCD * 2);
  _Float16* w3T   = (_Float16*)alloc((size_t)NP3 * FCD * 2);
  _Float16* mw2T  = (_Float16*)alloc((size_t)NPM * FCD * 2);
  float*    h1_32 = (float*)alloc((size_t)NROI * FCD * 4);
  _Float16* h1_16 = (_Float16*)alloc((size_t)NROI * FCD * 2);
  float*    h2_32 = (float*)alloc((size_t)NROI * FCD * 4);
  _Float16* h2_16 = (_Float16*)alloc((size_t)NROI * FCD * 2);
  float*    m1_32 = (float*)alloc((size_t)NROI * FCD * 4);
  _Float16* m1_16 = (_Float16*)alloc((size_t)NROI * FCD * 2);
  float*    of_32 = (float*)alloc((size_t)NROI * NP3 * 4);
  _Float16* of_16 = (_Float16*)alloc((size_t)NROI * NP3 * 2);
  float*    mk_32 = (float*)alloc((size_t)NROI * NPM * 4);
  _Float16* mk_16 = (_Float16*)alloc((size_t)NROI * NPM * 2);
  float*    b1p   = (float*)alloc((size_t)FCD * 4);
  float*    b2p   = (float*)alloc((size_t)FCD * 4);
  float*    b3p   = (float*)alloc((size_t)NP3 * 4);
  float*    mb1p  = (float*)alloc((size_t)FCD * 4);
  float*    mb2p  = (float*)alloc((size_t)NPM * 4);
  (void)ws_size; (void)in_sizes; (void)n_in; (void)out_size;

  // 1) weight / bias prep
  {
    size_t tot = (size_t)FCD * FIN;
    int blocks = (int)((tot + 255) / 256);
    cvt_weight<<<blocks, 256, 0, stream>>>(w1, w1T, FIN, FCD, FCD);
    cvt_weight<<<blocks, 256, 0, stream>>>(mw1, mw1T, FIN, FCD, FCD);
    cvt_weight<<<(FCD * FCD + 255) / 256, 256, 0, stream>>>(w2, w2T, FCD, FCD, FCD);
    cvt_weight<<<(NP3 * FCD + 255) / 256, 256, 0, stream>>>(w3, w3T, FCD, 98, NP3);
    cvt_weight<<<(NPM * FCD + 255) / 256, 256, 0, stream>>>(mw2, mw2T, FCD, 49, NPM);
    pad_bias<<<(FCD + 255) / 256, 256, 0, stream>>>(b1, b1p, FCD, FCD);
    pad_bias<<<(FCD + 255) / 256, 256, 0, stream>>>(b2, b2p, FCD, FCD);
    pad_bias<<<1, 256, 0, stream>>>(b3, b3p, 98, NP3);
    pad_bias<<<(FCD + 255) / 256, 256, 0, stream>>>(mb1, mb1p, FCD, FCD);
    pad_bias<<<1, 256, 0, stream>>>(mb2, mb2p, 49, NPM);
  }

  // 2) stage-1 pool -> xf16
  pool_kernel<<<NROI, CCH, 0, stream>>>(data, rois, nullptr, 0, nullptr, 0,
                                        xf16, nullptr, 0);

  // 3) FC stack (WMMA GEMMs); N-block = 128 cols per wave.
  gemm_wmma<<<dim3(NROI / 16, FCD / 128), 32, 0, stream>>>(
      xf16, w1T, b1p, h1_32, h1_16, NROI, FIN, FCD, /*relu*/0);
  gemm_wmma<<<dim3(NROI / 16, FCD / 128), 32, 0, stream>>>(
      h1_16, w2T, b2p, h2_32, h2_16, NROI, FCD, FCD, /*relu*/0);
  gemm_wmma<<<dim3(NROI / 16, NP3 / 128), 32, 0, stream>>>(
      h2_16, w3T, b3p, of_32, of_16, NROI, FCD, NP3, /*none*/1);
  gemm_wmma<<<dim3(NROI / 16, FCD / 128), 32, 0, stream>>>(
      xf16, mw1T, mb1p, m1_32, m1_16, NROI, FIN, FCD, /*relu*/0);
  gemm_wmma<<<dim3(NROI / 16, NPM / 128), 32, 0, stream>>>(
      m1_16, mw2T, mb2p, mk_32, mk_16, NROI, FCD, NPM, /*sigmoid*/2);

  // 4) stage-2 pool with offsets, scaled by mask -> final output
  pool_kernel<<<NROI, CCH, 0, stream>>>(data, rois, of_32, NP3, mk_32, NPM,
                                        nullptr, out, 1);
}
